// SwinTransformer3d_22144851378725
// MI455X (gfx1250) — matrix operand order repaired and checked by reading, not compiled, not run on hardware
//
#include <hip/hip_runtime.h>

typedef _Float16 half_t;
typedef __attribute__((ext_vector_type(16))) _Float16 v16h;
typedef __attribute__((ext_vector_type(8)))  _Float16 v8h;
typedef __attribute__((ext_vector_type(4)))  _Float16 v4h;
typedef __attribute__((ext_vector_type(8)))  float    v8f;

// ---- static problem config (matches reference) ----
constexpr int Bc   = 2;
constexpr int Tc   = 16;
constexpr int Hc   = 56;
constexpr int Wc   = 56;
constexpr int CDIM = 128;
constexpr int NHEAD= 4;
constexpr int HD   = 32;          // head dim
constexpr int VOL  = 392;         // 8*7*7 tokens per window
constexpr int VP   = 416;         // VOL padded to 13*32 (WMMA K granularity)
constexpr int QKV3 = 384;         // 3*C
constexpr int NWIN = 256;         // B * 2*8*8

// ---------------- WMMA helpers (CDNA5 16x16x32 f16 -> f32) ----------------
__device__ inline v8f wmma16(v16h a, v16h b, v8f c) {
  return __builtin_amdgcn_wmma_f32_16x16x32_f16(false, a, false, b, (short)0, c,
                                                false, false);
}

// A fragment: 16x32 (MxK) from row-major f16 [.. , ld]; lane l<16 holds row M=l,
// K = {0..15}; lane l>=16 holds K = {16..31} (matches B fragment order).
__device__ inline v16h load_a_frag(const half_t* A, int ld, int mbase, int kbase,
                                   int lane) {
  int r = lane & 15, h = lane >> 4;
  const half_t* p = A + (mbase + r) * ld + kbase + h * 16;
  union { v16h v; v8h c[2]; } u;
  u.c[0] = *(const v8h*)p;
  u.c[1] = *(const v8h*)(p + 8);
  return u.v;
}

// B fragment: 32x16 (KxN) loaded from "BT" storage = row-major [N][K]:
// lane l holds column N = l&15, contiguous K = (l>>4)*16 .. +15  -> one 32B load.
__device__ inline v16h load_bt_frag(const half_t* BT, int ld, int nbase, int kbase,
                                    int lane) {
  const half_t* p = BT + (nbase + (lane & 15)) * ld + kbase + (lane >> 4) * 16;
  return *(const v16h*)p;
}

__device__ inline v4h pack4(float a, float b, float c, float d) {
  v4h r; r[0] = (_Float16)a; r[1] = (_Float16)b; r[2] = (_Float16)c; r[3] = (_Float16)d;
  return r;
}

// --------- CDNA5 async global->LDS copy (ASYNCcnt-tracked DMA path) --------
// VDST = LDS byte offset (low 32 bits of generic shared pointer; flat-LDS
// aperture maps addr[31:0] -> wave-relative LDS offset), VADDR = 64-bit global.
__device__ inline void async_copy_b128(void* lds_dst, const void* gsrc) {
  uint32_t la = (uint32_t)(uintptr_t)lds_dst;
  uint64_t ga = (uint64_t)(uintptr_t)gsrc;
  asm volatile("global_load_async_to_lds_b128 %0, %1, off"
               :: "v"(la), "v"(ga) : "memory");
}
__device__ inline void wait_async0() {
  asm volatile("s_wait_asynccnt 0x0" ::: "memory");
}

// shifted-window region id for the attention mask (recomputed on device)
__device__ inline int region_id(int nt_i, int nh_i, int nw_i, int v) {
  int t_l = v / 49, hw = v % 49, h_l = hw / 7, w_l = hw % 7;
  int ts = nt_i * 8 + t_l, hs = nh_i * 7 + h_l, ws = nw_i * 7 + w_l;
  int tr = (ts < 8)  ? 0 : (ts < 12 ? 1 : 2);   // [0,T-8),[T-8,T-4),[T-4,T)
  int hr = (hs < 49) ? 0 : (hs < 53 ? 1 : 2);
  int wr = (ws < 49) ? 0 : (ws < 53 ? 1 : 2);
  return tr * 9 + hr * 3 + wr;
}

// ============================ Kernel 1: QKV proj ============================
// one workgroup (8 waves) per window
__global__ void k_qkv(const float* __restrict__ x, const float* __restrict__ w,
                      const float* __restrict__ bias,
                      half_t* __restrict__ Qs, half_t* __restrict__ Ks,
                      half_t* __restrict__ VT) {
  extern __shared__ char smem[];
  half_t* Ax = (half_t*)smem;          // [VP][128] f16
  half_t* Wl = Ax + VP * CDIM;         // [384][128] f16

  const int tid = threadIdx.x;
  const int win = blockIdx.x;
  const int b = win >> 7, rem = win & 127;
  const int nt_i = rem >> 6, nh_i = (rem >> 3) & 7, nw_i = rem & 7;

  // stage X window (shifted gather), f32 -> f16, zero-pad rows >= VOL
  for (int i = tid; i < VP * (CDIM / 4); i += 256) {
    int v = i >> 5, c4 = (i & 31) * 4;
    v4h h = pack4(0.f, 0.f, 0.f, 0.f);
    if (v < VOL) {
      int t_l = v / 49, hw = v % 49, h_l = hw / 7, w_l = hw % 7;
      int t = (nt_i * 8 + t_l + 4) & 15;
      int hh = nh_i * 7 + h_l + 3; if (hh >= 56) hh -= 56;
      int ww = nw_i * 7 + w_l + 3; if (ww >= 56) ww -= 56;
      const float4 f =
          *(const float4*)(x + ((((size_t)(b * Tc + t) * Hc + hh) * Wc + ww) * CDIM + c4));
      h = pack4(f.x, f.y, f.z, f.w);
    }
    *(v4h*)(Ax + v * CDIM + c4) = h;
  }
  // stage qkv weight f32 -> f16
  for (int i = tid; i < QKV3 * (CDIM / 4); i += 256) {
    int row = i >> 5, c4 = (i & 31) * 4;
    const float4 f = *(const float4*)(w + (size_t)row * CDIM + c4);
    *(v4h*)(Wl + row * CDIM + c4) = pack4(f.x, f.y, f.z, f.w);
  }
  __syncthreads();

  const int lane = tid & 31, wid = tid >> 5;
  const float rscale = 0.17677669529663687f;  // hd^-0.5

  for (int tile = wid; tile < 26 * 24; tile += 8) {
    int mt = tile / 24, nt = tile % 24;
    v8f acc = {};
#pragma unroll
    for (int kt = 0; kt < 4; ++kt) {
      v16h a  = load_a_frag(Ax, CDIM, mt * 16, kt * 32, lane);
      v16h bb = load_bt_frag(Wl, CDIM, nt * 16, kt * 32, lane);
      acc = wmma16(a, bb, acc);
    }
    int n = nt * 16 + (lane & 15);
    int which = n >> 7, head = (n >> 5) & 3, d = n & 31;
    float bn = bias[n];
    size_t base = (size_t)(win * NHEAD + head);
#pragma unroll
    for (int r = 0; r < 8; ++r) {
      int m = mt * 16 + (lane >> 4) * 8 + r;
      float val = (m < VOL) ? (acc[r] + bn) : 0.f;   // zero-pad tokens
      if (which == 0)      Qs[(base * VP + m) * HD + d] = (_Float16)(val * rscale);
      else if (which == 1) Ks[(base * VP + m) * HD + d] = (_Float16)val;
      else                 VT[(base * HD + d) * VP + m] = (_Float16)val;
    }
  }
}

// ====================== Kernel 2: windowed attention =======================
// one workgroup (8 waves) per (window, head); each wave owns 16-row M tiles
__global__ void k_attn(const half_t* __restrict__ Qs, const half_t* __restrict__ Ks,
                       const half_t* __restrict__ VT, const float* __restrict__ rpb,
                       half_t* __restrict__ AO) {
  extern __shared__ char smem[];
  half_t* Ql = (half_t*)smem;                 // [VP][32]
  half_t* Kl = Ql + VP * HD;                  // [VP][32]
  half_t* Vl = Kl + VP * HD;                  // [32][VP]  (V transposed)
  float*  S  = (float*)(Vl + HD * VP);        // 8 waves x [16][VP] f32
  int*    g  = (int*)(S + 8 * 16 * VP);       // [VP] region ids
  float* invs = (float*)(g + VP);             // 8 x 16 inverse row sums

  const int tid = threadIdx.x, lane = tid & 31, wid = tid >> 5;
  const int wi = blockIdx.x >> 2, head = blockIdx.x & 3;
  const int rem = wi & 127;
  const int nt_i = rem >> 6, nh_i = (rem >> 3) & 7, nw_i = rem & 7;

  // async DMA Q/K/Vt straight into LDS (no VGPR round-trip)
  const size_t hb = (size_t)(wi * NHEAD + head) * VP * HD;
  const uint4* qg = (const uint4*)(Qs + hb);
  const uint4* kg = (const uint4*)(Ks + hb);
  const uint4* vg = (const uint4*)(VT + hb);
  for (int i = tid; i < VP * HD / 8; i += 256) {
    async_copy_b128((uint4*)Ql + i, qg + i);
    async_copy_b128((uint4*)Kl + i, kg + i);
    async_copy_b128((uint4*)Vl + i, vg + i);
  }
  // overlap region-id computation with the in-flight async transfers
  for (int v = tid; v < VP; v += 256) g[v] = (v < VOL) ? region_id(nt_i, nh_i, nw_i, v) : -1;
  wait_async0();
  __syncthreads();

  const int n0 = lane & 15, hf = lane >> 4;
  const float* rpb_h = rpb + (size_t)head * VOL * VOL;

  for (int mt = wid; mt < 26; mt += 8) {
    float* Srow = S + wid * 16 * VP;
    v16h aq = load_a_frag(Ql, HD, mt * 16, 0, lane);   // single K-step (hd=32)

    for (int nt = 0; nt < 26; ++nt) {
      v16h bk = load_bt_frag(Kl, HD, nt * 16, 0, lane);
      v8f z = {};
      v8f acc = wmma16(aq, bk, z);
      int n = nt * 16 + n0;
#pragma unroll
      for (int r = 0; r < 8; ++r) {
        int ml = hf * 8 + r, m = mt * 16 + ml;
        float s = -1e30f;
        if (m < VOL && n < VOL)
          s = acc[r] + rpb_h[(size_t)m * VOL + n] + ((g[m] != g[n]) ? -100.f : 0.f);
        Srow[ml * VP + n] = s;
      }
    }
    asm volatile("s_wait_dscnt 0x0" ::: "memory");

    // wave-cooperative softmax over 16 rows; write unnormalized P as f16 in-place
    half_t* P = (half_t*)Srow;
    for (int row = 0; row < 16; ++row) {
      float* p = Srow + row * VP;
      float mx = -3.4e38f;
      for (int c = lane; c < VP; c += 32) mx = fmaxf(mx, p[c]);
      for (int o = 16; o > 0; o >>= 1) mx = fmaxf(mx, __shfl_xor(mx, o, 32));
      float sum = 0.f;
      for (int c = lane; c < VP; c += 32) {        // read f32 then overwrite as f16:
        float e = __expf(p[c] - mx);               // f16 row r only clobbers f32 row r/2
        sum += e;
        P[row * VP + c] = (_Float16)e;
      }
      for (int o = 16; o > 0; o >>= 1) sum += __shfl_xor(sum, o, 32);
      if (lane == 0) invs[wid * 16 + row] = 1.f / sum;
    }
    asm volatile("s_wait_dscnt 0x0" ::: "memory");

    // out = P @ V  (K = 13 x 32), fold 1/rowsum in epilogue
    for (int nt2 = 0; nt2 < 2; ++nt2) {
      v8f acc = {};
#pragma unroll
      for (int kt = 0; kt < 13; ++kt) {
        v16h pa = load_a_frag(P, VP, 0, kt * 32, lane);
        v16h vb = load_bt_frag(Vl, VP, nt2 * 16, kt * 32, lane);
        acc = wmma16(pa, vb, acc);
      }
#pragma unroll
      for (int r = 0; r < 8; ++r) {
        int ml = hf * 8 + r, m = mt * 16 + ml;
        float o = (m < VOL) ? acc[r] * invs[wid * 16 + ml] : 0.f;
        AO[((size_t)wi * VP + m) * CDIM + head * HD + nt2 * 16 + n0] = (_Float16)o;
      }
    }
  }
}

// =================== Kernel 3: output proj + un-shift scatter ==============
__global__ void k_proj(const half_t* __restrict__ AO, const float* __restrict__ w,
                       const float* __restrict__ bias, float* __restrict__ out) {
  extern __shared__ char smem[];
  half_t* Al = (half_t*)smem;            // [VP][128]
  half_t* Wl = Al + VP * CDIM;           // [128][128]

  const int tid = threadIdx.x, win = blockIdx.x;
  const int b = win >> 7, rem = win & 127;
  const int nt_i = rem >> 6, nh_i = (rem >> 3) & 7, nw_i = rem & 7;

  // async DMA the attention output tile into LDS
  const uint4* src = (const uint4*)(AO + (size_t)win * VP * CDIM);
  for (int i = tid; i < VP * CDIM / 8; i += 256)
    async_copy_b128((uint4*)Al + i, src + i);
  // overlap: convert proj weight f32 -> f16 while the DMA is in flight
  for (int i = tid; i < CDIM * (CDIM / 4); i += 256) {
    int row = i >> 5, c4 = (i & 31) * 4;
    const float4 f = *(const float4*)(w + (size_t)row * CDIM + c4);
    *(v4h*)(Wl + row * CDIM + c4) = pack4(f.x, f.y, f.z, f.w);
  }
  wait_async0();
  __syncthreads();

  const int lane = tid & 31, wid = tid >> 5;
  for (int tile = wid; tile < 26 * 8; tile += 8) {
    int mt = tile >> 3, nt = tile & 7;
    v8f acc = {};
#pragma unroll
    for (int kt = 0; kt < 4; ++kt) {
      v16h a  = load_a_frag(Al, CDIM, mt * 16, kt * 32, lane);
      v16h bb = load_bt_frag(Wl, CDIM, nt * 16, kt * 32, lane);
      acc = wmma16(a, bb, acc);
    }
    int n = nt * 16 + (lane & 15);
    float bn = bias[n];
#pragma unroll
    for (int r = 0; r < 8; ++r) {
      int m = mt * 16 + (lane >> 4) * 8 + r;
      if (m < VOL) {
        int t_l = m / 49, hw = m % 49, h_l = hw / 7, w_l = hw % 7;
        int t = (nt_i * 8 + t_l + 4) & 15;                   // roll-back == gather map
        int hh = nh_i * 7 + h_l + 3; if (hh >= 56) hh -= 56;
        int ww = nw_i * 7 + w_l + 3; if (ww >= 56) ww -= 56;
        out[(((size_t)(b * Tc + t) * Hc + hh) * Wc + ww) * CDIM + n] = acc[r] + bn;
      }
    }
  }
}

// ================================ launcher =================================
extern "C" void kernel_launch(void* const* d_in, const int* in_sizes, int n_in,
                              void* d_out, int out_size, void* d_ws, size_t ws_size,
                              hipStream_t stream) {
  const float* x     = (const float*)d_in[0];
  const float* qkvw  = (const float*)d_in[1];
  const float* qkvb  = (const float*)d_in[2];
  const float* projw = (const float*)d_in[3];
  const float* projb = (const float*)d_in[4];
  const float* rpb   = (const float*)d_in[5];
  float* out = (float*)d_out;

  const size_t per = (size_t)NWIN * NHEAD * VP * HD;   // 13,631,488 halves
  half_t* Qs = (half_t*)d_ws;
  half_t* Ks = Qs + per;
  half_t* VT = Ks + per;
  half_t* AO = VT + per;                               // NWIN*VP*CDIM == per

  const size_t smem1 = (size_t)(VP * CDIM + QKV3 * CDIM) * sizeof(half_t);          // 200 KB
  const size_t smem2 = (size_t)(3 * VP * HD) * sizeof(half_t) +
                       (size_t)(8 * 16 * VP) * sizeof(float) +
                       (size_t)VP * sizeof(int) + (size_t)(8 * 16) * sizeof(float); // ~288 KB
  const size_t smem3 = (size_t)(VP * CDIM + CDIM * CDIM) * sizeof(half_t);          // 136 KB

  k_qkv <<<dim3(NWIN),        dim3(256), smem1, stream>>>(x, qkvw, qkvb, Qs, Ks, VT);
  k_attn<<<dim3(NWIN * NHEAD),dim3(256), smem2, stream>>>(Qs, Ks, VT, rpb, AO);
  k_proj<<<dim3(NWIN),        dim3(256), smem3, stream>>>(AO, projw, projb, out);
}
